// Model_49280454754500
// MI455X (gfx1250) — matrix operand order repaired and checked by reading, not compiled, not run on hardware
//
#include <hip/hip_runtime.h>
#include <hip/hip_fp16.h>
#include <stdint.h>

typedef __attribute__((ext_vector_type(16))) _Float16 v16h;
typedef __attribute__((ext_vector_type(8)))  _Float16 v8h;
typedef __attribute__((ext_vector_type(4)))  _Float16 v4h;
typedef __attribute__((ext_vector_type(2)))  _Float16 v2h;
typedef __attribute__((ext_vector_type(8)))  float    v8f;

#define B_ROWS 16384
#define KFEAT  32
#define NFTS   40960
#define HDIM   256

__device__ __forceinline__ float clip01(float x) { return fminf(fmaxf(x, 0.f), 1.f); }

// ---------------------------------------------------------------------------
// Kernel 0a: convert ft_w (f32, 40960x256) -> f16 table (halves L2 gather bytes)
// ---------------------------------------------------------------------------
__global__ __launch_bounds__(256)
void cvt_ftw_kernel(const float* __restrict__ src, _Float16* __restrict__ dst, int n4) {
  int i = blockIdx.x * blockDim.x + threadIdx.x;
  if (i >= n4) return;
  const float4 f = ((const float4*)src)[i];
  v4h h = { (_Float16)f.x, (_Float16)f.y, (_Float16)f.z, (_Float16)f.w };
  ((v4h*)dst)[i] = h;
}

// ---------------------------------------------------------------------------
// Kernel 0b: convert fc1_w (32x512) and fc2_w (32x32) to f16 (row-major, n-major)
// ---------------------------------------------------------------------------
__global__ __launch_bounds__(256)
void cvt_small_kernel(const float* __restrict__ fc1w, const float* __restrict__ fc2w,
                      _Float16* __restrict__ fc1h, _Float16* __restrict__ fc2h) {
  const int i = blockIdx.x * blockDim.x + threadIdx.x;
  const int stride = gridDim.x * blockDim.x;
  for (int j = i; j < 32 * 512; j += stride) fc1h[j] = (_Float16)fc1w[j];
  for (int j = i; j < 32 * 32;  j += stride) fc2h[j] = (_Float16)fc2w[j];
}

// ---------------------------------------------------------------------------
// Kernel 1: feature transformer. 128 threads serve one batch row (2 rows/block).
// Thread tl owns H elements (2*tl, 2*tl+1); gathers are 128x4B = 512B coalesced
// per feature row, hitting L2 (table resident). Index/val staged in LDS; the
// idx>=0 branch is wave-uniform (whole wave shares one batch row).
// ---------------------------------------------------------------------------
template <bool HT>
__global__ __launch_bounds__(256)
void ft_kernel(const int* __restrict__ wic, const float* __restrict__ wval,
               const int* __restrict__ bic, const float* __restrict__ bval,
               const int* __restrict__ stm, const float* __restrict__ ftw,
               const _Float16* __restrict__ ftwh, const float* __restrict__ ftb,
               _Float16* __restrict__ xh) {
  __shared__ int   sIdx[2][64];
  __shared__ float sVal[2][64];
  const int t  = threadIdx.x;
  const int rp = t >> 7;        // which of the 2 batch rows in this block
  const int tl = t & 127;       // H-pair index
  const int b  = blockIdx.x * 2 + rp;

  if (tl < 32) {
    sIdx[rp][tl] = wic[b * KFEAT + tl];
    sVal[rp][tl] = wval[b * KFEAT + tl];
  } else if (tl < 64) {
    sIdx[rp][tl] = bic[b * KFEAT + (tl - 32)];
    sVal[rp][tl] = bval[b * KFEAT + (tl - 32)];
  }
  __syncthreads();

  float w0 = 0.f, w1 = 0.f, c0 = 0.f, c1 = 0.f;
#pragma unroll 8
  for (int k = 0; k < 32; ++k) {           // white features
    const int idx = sIdx[rp][k];
    if (idx >= 0) {
      const float v = sVal[rp][k];
      if (HT) {
        const v2h p = ((const v2h*)(ftwh + (size_t)idx * HDIM))[tl];
        w0 += v * (float)p[0]; w1 += v * (float)p[1];
      } else {
        const float2 p = ((const float2*)(ftw + (size_t)idx * HDIM))[tl];
        w0 += v * p.x; w1 += v * p.y;
      }
    }
  }
#pragma unroll 8
  for (int k = 32; k < 64; ++k) {          // black features
    const int idx = sIdx[rp][k];
    if (idx >= 0) {
      const float v = sVal[rp][k];
      if (HT) {
        const v2h p = ((const v2h*)(ftwh + (size_t)idx * HDIM))[tl];
        c0 += v * (float)p[0]; c1 += v * (float)p[1];
      } else {
        const float2 p = ((const float2*)(ftw + (size_t)idx * HDIM))[tl];
        c0 += v * p.x; c1 += v * p.y;
      }
    }
  }

  const float2 bb = ((const float2*)ftb)[tl];
  const float wf0 = w0 + bb.x, wf1 = w1 + bb.y;
  const float bf0 = c0 + bb.x, bf1 = c1 + bb.y;
  const bool swap = stm[b] != 0;
  const float f0 = clip01(swap ? bf0 : wf0), f1 = clip01(swap ? bf1 : wf1);
  const float g0 = clip01(swap ? wf0 : bf0), g1 = clip01(swap ? wf1 : bf1);

  v2h oA = { (_Float16)f0, (_Float16)f1 };
  v2h oB = { (_Float16)g0, (_Float16)g1 };
  ((v2h*)(xh + (size_t)b * 512))[tl]       = oA;   // first 256
  ((v2h*)(xh + (size_t)b * 512 + 256))[tl] = oB;   // second 256
}

// ---------------------------------------------------------------------------
// Kernel 2: fused FC1 -> FC2 -> FCO. One wave per 16-row batch tile.
// WMMA f32_16x16x32_f16. Lane mapping used here (A/B k order is this code's contiguous choice; C/D per ISA):
//   A: M = lane&15, g = lane>>4; halves h -> K = 32c + 16g + h (contiguous)
//   B: N = lane&15; halves h -> K = 32c + 16g + h (contiguous -> 32B load of W[n][k])
//   C/D: element r -> (m = r + 8g, n = lane&15)
// ---------------------------------------------------------------------------
__global__ __launch_bounds__(256)
void fc_kernel(const _Float16* __restrict__ xh,  const _Float16* __restrict__ fc1h,
               const float* __restrict__ fc1b,   const _Float16* __restrict__ fc2h,
               const float* __restrict__ fc2b,   const float* __restrict__ fcow,
               const float* __restrict__ fcob,   float* __restrict__ out) {
  __shared__ _Float16 sAct[8 * 16 * 32];   // per-wave 16x32 f16 act1 staging

  const int lane = threadIdx.x & 31;
  const int wave = threadIdx.x >> 5;
  const int tile = blockIdx.x * 8 + wave;
  const int g    = lane >> 4;
  const int nl   = lane & 15;
  const int m0   = tile * 16;

  // ---- FC1: [16,512] x [512,32] via 2 N-tiles x 16 K-chunks ----
  v8f acc0 = {}; v8f acc1 = {};
  const _Float16* xrow = xh  + (size_t)(m0 + nl) * 512;
  const _Float16* w0p  = fc1h + (size_t)nl * 512;          // n = nl
  const _Float16* w1p  = fc1h + (size_t)(16 + nl) * 512;   // n = 16+nl
#pragma unroll 4
  for (int c = 0; c < 16; ++c) {
    const int ko = c * 32;
    const v8h alo = *(const v8h*)(xrow + ko + g * 16);
    const v8h ahi = *(const v8h*)(xrow + ko + g * 16 + 8);
    v16h a;
#pragma unroll
    for (int i = 0; i < 8; ++i) { a[i] = alo[i]; a[i + 8] = ahi[i]; }
    const v16h b0 = *(const v16h*)(w0p + ko + g * 16);
    const v16h b1 = *(const v16h*)(w1p + ko + g * 16);
    acc0 = __builtin_amdgcn_wmma_f32_16x16x32_f16(false, a, false, b0, (short)0, acc0, false, false);
    acc1 = __builtin_amdgcn_wmma_f32_16x16x32_f16(false, a, false, b1, (short)0, acc1, false, false);
  }

  // ---- bias + clip, stage act1 (16x32) row-major in this wave's LDS slice ----
  const float bi0 = fc1b[nl], bi1 = fc1b[16 + nl];
  _Float16* wact = sAct + wave * 512;
#pragma unroll
  for (int r = 0; r < 8; ++r) {
    const int m = r + g * 8;
    wact[m * 32 + nl]      = (_Float16)clip01(acc0[r] + bi0);
    wact[m * 32 + 16 + nl] = (_Float16)clip01(acc1[r] + bi1);
  }
  __builtin_amdgcn_wave_barrier();   // DS is in-order per wave; stop compiler reordering

  // ---- FC2: [16,32] x [32,32] ----
  const v8h a2lo = *(const v8h*)(wact + nl * 32 + g * 16);
  const v8h a2hi = *(const v8h*)(wact + nl * 32 + g * 16 + 8);
  v16h a2;
#pragma unroll
  for (int i = 0; i < 8; ++i) { a2[i] = a2lo[i]; a2[i + 8] = a2hi[i]; }
  const v16h b20 = *(const v16h*)(fc2h + nl * 32 + g * 16);
  const v16h b21 = *(const v16h*)(fc2h + (16 + nl) * 32 + g * 16);
  v8f d0 = {}; v8f d1 = {};
  d0 = __builtin_amdgcn_wmma_f32_16x16x32_f16(false, a2, false, b20, (short)0, d0, false, false);
  d1 = __builtin_amdgcn_wmma_f32_16x16x32_f16(false, a2, false, b21, (short)0, d1, false, false);

  // ---- FCO: per-row dot over 32 cols, reduce across the 16 lanes of each half ----
  const float b2lo = fc2b[nl], b2hi = fc2b[16 + nl];
  const float wlo  = fcow[nl], whi  = fcow[16 + nl];
  const float ob   = fcob[0];
#pragma unroll
  for (int r = 0; r < 8; ++r) {
    float p = clip01(d0[r] + b2lo) * wlo + clip01(d1[r] + b2hi) * whi;
    p += __shfl_xor(p, 8, 32);
    p += __shfl_xor(p, 4, 32);
    p += __shfl_xor(p, 2, 32);
    p += __shfl_xor(p, 1, 32);
    if (nl == 0) out[m0 + r + g * 8] = p + ob;
  }
}

// ---------------------------------------------------------------------------
extern "C" void kernel_launch(void* const* d_in, const int* in_sizes, int n_in,
                              void* d_out, int out_size, void* d_ws, size_t ws_size,
                              hipStream_t stream) {
  const int*   wic  = (const int*)  d_in[0];
  const float* wval = (const float*)d_in[1];
  const int*   bic  = (const int*)  d_in[2];
  const float* bval = (const float*)d_in[3];
  const int*   stm  = (const int*)  d_in[4];
  const float* ftw  = (const float*)d_in[5];
  const float* ftb  = (const float*)d_in[6];
  const float* fc1w = (const float*)d_in[7];
  const float* fc1b = (const float*)d_in[8];
  const float* fc2w = (const float*)d_in[9];
  const float* fc2b = (const float*)d_in[10];
  const float* fcow = (const float*)d_in[11];
  const float* fcob = (const float*)d_in[12];
  float* out = (float*)d_out;

  // Scratch layout (256B aligned chunks)
  const size_t ftwh_bytes = (size_t)NFTS * HDIM * 2;        // 20.97 MB
  const size_t xh_bytes   = (size_t)B_ROWS * 512 * 2;       // 16.78 MB
  const size_t small_bytes = ((32 * 512 + 32 * 32) * 2 + 511) & ~255ull;
  const bool useHT = ws_size >= ftwh_bytes + xh_bytes + small_bytes + 4096;

  char* ws = (char*)d_ws;
  size_t off = 0;
  auto carve = [&](size_t bytes) { void* p = ws + off; off += (bytes + 255) & ~255ull; return p; };

  _Float16* ftwh = nullptr;
  if (useHT) ftwh = (_Float16*)carve(ftwh_bytes);
  _Float16* xh   = (_Float16*)carve(xh_bytes);
  _Float16* fc1h = (_Float16*)carve(32 * 512 * 2);
  _Float16* fc2h = (_Float16*)carve(32 * 32 * 2);

  if (useHT) {
    const int n4 = NFTS * HDIM / 4;                         // 2,621,440
    cvt_ftw_kernel<<<n4 / 256, 256, 0, stream>>>(ftw, ftwh, n4);
  }
  cvt_small_kernel<<<64, 256, 0, stream>>>(fc1w, fc2w, fc1h, fc2h);

  if (useHT) {
    ft_kernel<true><<<B_ROWS / 2, 256, 0, stream>>>(wic, wval, bic, bval, stm,
                                                    ftw, ftwh, ftb, xh);
  } else {
    ft_kernel<false><<<B_ROWS / 2, 256, 0, stream>>>(wic, wval, bic, bval, stm,
                                                     ftw, ftwh, ftb, xh);
  }

  fc_kernel<<<B_ROWS / (16 * 8), 256, 0, stream>>>(xh, fc1h, fc1b, fc2h, fc2b,
                                                   fcow, fcob, out);
}
